// MoELayer_54348516163739
// MI455X (gfx1250) — matrix operand order repaired and checked by reading, not compiled, not run on hardware
//
#include <hip/hip_runtime.h>
#include <hip/hip_bf16.h>
#include <stdint.h>
#include <stddef.h>

#define N_TOK  131072
#define D_IN   256
#define D_HID  512
#define D_OUT  256
#define NEXP   8
#define LN_EPS 1e-5f
#define TM     32            // tokens per FFN block (2 x 16-row WMMA tiles)

typedef __bf16 bf16;
typedef __attribute__((ext_vector_type(16))) __bf16 v16bf;
typedef __attribute__((ext_vector_type(8)))  __bf16 v8bf;
typedef __attribute__((ext_vector_type(8)))  float  v8f;

// ---------------- workspace layout (bytes) ----------------
#define WS_XB   ((size_t)0)                                   // N*256 bf16
#define WS_W1T  (WS_XB  + (size_t)N_TOK*D_IN*2)               // E*512*256 bf16 (n-major)
#define WS_W2T  (WS_W1T + (size_t)NEXP*D_HID*D_IN*2)          // E*512*512 bf16 (n-major)
#define WS_W3T  (WS_W2T + (size_t)NEXP*D_HID*D_HID*2)         // E*256*512 bf16 (n-major)
#define WS_CNT  (WS_W3T + (size_t)NEXP*D_OUT*D_HID*2)         // E ints (padded 256B)
#define WS_IDX  (WS_CNT + 256)                                // E*N ints
#define WS_WGT  (WS_IDX + (size_t)NEXP*N_TOK*4)               // E*N floats

static __device__ __forceinline__ bf16 f2bf(float f) {
    union { float f; uint32_t u; } v; v.f = f;
    uint32_t r = v.u + 0x7FFFu + ((v.u >> 16) & 1u);   // RNE
    uint16_t h = (uint16_t)(r >> 16);
    bf16 b; __builtin_memcpy(&b, &h, 2);
    return b;
}
static __device__ __forceinline__ uint16_t f2bfbits(float f) {
    union { float f; uint32_t u; } v; v.f = f;
    uint32_t r = v.u + 0x7FFFu + ((v.u >> 16) & 1u);
    return (uint16_t)(r >> 16);
}
static __device__ __forceinline__ float gelu_exact(float x) {
    return 0.5f * x * (1.0f + erff(x * 0.70710678118654752f));
}

// ---------------- kernel 1: weight convert + transpose to N-major bf16 ----------------
#define R1 (NEXP*D_HID*D_IN)    // 1048576
#define R2 (NEXP*D_HID*D_HID)   // 2097152
#define R3 (NEXP*D_OUT*D_HID)   // 1048576

__global__ void moe_prep_w(const float* __restrict__ W1, const float* __restrict__ W2,
                           const float* __restrict__ W3,
                           bf16* __restrict__ w1t, bf16* __restrict__ w2t,
                           bf16* __restrict__ w3t) {
    int i = blockIdx.x * blockDim.x + threadIdx.x;
    if (i < R1) {                       // w1t[e][n*256+k] = W1[e][k*512+n]
        int e = i >> 17, r = i & 131071, n = r >> 8, k = r & 255;
        w1t[i] = f2bf(W1[(size_t)e * (D_IN*D_HID) + (size_t)k * D_HID + n]);
    } else if (i < R1 + R2) {           // w2t[e][n*512+k] = W2[e][k*512+n]
        int j = i - R1;
        int e = j >> 18, r = j & 262143, n = r >> 9, k = r & 511;
        w2t[j] = f2bf(W2[(size_t)e * (D_HID*D_HID) + (size_t)k * D_HID + n]);
    } else {                            // w3t[e][n*512+k] = W3[e][k*256+n]
        int j = i - R1 - R2;
        int e = j >> 17, r = j & 131071, n = r >> 9, k = r & 511;
        w3t[j] = f2bf(W3[(size_t)e * (D_HID*D_OUT) + (size_t)k * D_OUT + n]);
    }
}

// ---------------- kernel 2: zero output + routing counters ----------------
__global__ void moe_zero(float* __restrict__ out, int* __restrict__ counts) {
    size_t i = (size_t)blockIdx.x * blockDim.x + threadIdx.x;
    if (i < (size_t)N_TOK * D_OUT) out[i] = 0.0f;
    if (i < NEXP) counts[i] = 0;
}

// ---------------- kernel 3: gating + routing + x -> bf16 ----------------
// one wave32 per token
__global__ void moe_gate(const float* __restrict__ x, const float* __restrict__ gw1,
                         const float* __restrict__ gw2, bf16* __restrict__ xb,
                         int* __restrict__ counts, int* __restrict__ idxlist,
                         float* __restrict__ wlist) {
    int tok  = (int)((blockIdx.x * blockDim.x + threadIdx.x) >> 5);
    int lane = threadIdx.x & 31;
    if (tok >= N_TOK) return;
    const float* xr = x + (size_t)tok * D_IN;

    float xv[8];
    float acc[16];
#pragma unroll
    for (int c = 0; c < 16; ++c) acc[c] = 0.0f;
#pragma unroll
    for (int i = 0; i < 8; ++i) {
        int d = lane + 32 * i;
        float v = xr[d];
        xv[i] = v;
        const float* g = gw1 + (size_t)d * 16;
#pragma unroll
        for (int c = 0; c < 16; ++c) acc[c] += v * g[c];
    }
    // butterfly reduce across wave32
#pragma unroll
    for (int off = 16; off >= 1; off >>= 1) {
#pragma unroll
        for (int c = 0; c < 16; ++c) acc[c] += __shfl_xor(acc[c], off, 32);
    }
    // store bf16 copy of x (coalesced)
#pragma unroll
    for (int i = 0; i < 8; ++i) xb[(size_t)tok * D_IN + lane + 32 * i] = f2bf(xv[i]);

    if (lane == 0) {
        float th[16];
#pragma unroll
        for (int c = 0; c < 16; ++c) th[c] = tanhf(acc[c]);
        float lg[NEXP];
#pragma unroll
        for (int j = 0; j < NEXP; ++j) {
            float s = 0.0f;
#pragma unroll
            for (int c = 0; c < 16; ++c) s += th[c] * gw2[c * NEXP + j];
            lg[j] = s;
        }
        float mx = lg[0];
#pragma unroll
        for (int j = 1; j < NEXP; ++j) mx = fmaxf(mx, lg[j]);
        float pr[NEXP]; float sum = 0.0f;
#pragma unroll
        for (int j = 0; j < NEXP; ++j) { pr[j] = expf(lg[j] - mx); sum += pr[j]; }
        float inv = 1.0f / sum;
#pragma unroll
        for (int j = 0; j < NEXP; ++j) pr[j] *= inv;
        int i1 = 0;
#pragma unroll
        for (int j = 1; j < NEXP; ++j) if (pr[j] > pr[i1]) i1 = j;
        int i2 = (i1 == 0) ? 1 : 0;
#pragma unroll
        for (int j = 0; j < NEXP; ++j) if (j != i1 && pr[j] > pr[i2]) i2 = j;
        float w1 = pr[i1], w2 = pr[i2];
        float nrm = 1.0f / (w1 + w2 + 1e-12f);
        w1 *= nrm; w2 *= nrm;
        int s1 = atomicAdd(&counts[i1], 1);
        idxlist[(size_t)i1 * N_TOK + s1] = tok; wlist[(size_t)i1 * N_TOK + s1] = w1;
        int s2 = atomicAdd(&counts[i2], 1);
        idxlist[(size_t)i2 * N_TOK + s2] = tok; wlist[(size_t)i2 * N_TOK + s2] = w2;
    }
}

// ---------------- LN + GeLU over a 32x512 fp32 tile in LDS -> bf16 LDS ----------------
// 256 threads: 8 threads per row, 64 columns each
static __device__ __forceinline__ void ln_gelu_rows(float* HF, unsigned short* HB,
                                                    const float* __restrict__ gg,
                                                    const float* __restrict__ bb,
                                                    float* red1, float* red2,
                                                    float* smu, float* srs, int tid) {
    __syncthreads();                   // HF fully written
    int r = tid >> 3, seg = tid & 7;   // r<32, seg<8
    float s = 0.0f, ss = 0.0f;
    for (int c = seg * 64; c < seg * 64 + 64; ++c) {
        float h = HF[r * D_HID + c];
        s += h; ss += h * h;
    }
    red1[r * 8 + seg] = s; red2[r * 8 + seg] = ss;
    __syncthreads();
    if (tid < TM) {
        float a = 0.0f, b = 0.0f;
#pragma unroll
        for (int i = 0; i < 8; ++i) { a += red1[tid * 8 + i]; b += red2[tid * 8 + i]; }
        float mu = a * (1.0f / D_HID);
        float var = b * (1.0f / D_HID) - mu * mu;
        smu[tid] = mu; srs[tid] = rsqrtf(var + LN_EPS);
    }
    __syncthreads();
    float mu = smu[r], rs = srs[r];
    for (int c = seg * 64; c < seg * 64 + 64; ++c) {
        float h = (HF[r * D_HID + c] - mu) * rs * gg[c] + bb[c];
        HB[r * D_HID + c] = f2bfbits(gelu_exact(h));
    }
    __syncthreads();
}

// ---------------- kernel 4: fused 3-stage expert FFN on a 32-token tile ----------------
// grid: (N_TOK/32, NEXP), block: 256 (8 wave32)
// Each B fragment is reused by 2 WMMAs (two 16-row M-tiles) -> 2x arithmetic
// intensity per weight byte vs. the 16-token version.
__global__ void __launch_bounds__(256)
moe_ffn(const bf16* __restrict__ xb,
        const bf16* __restrict__ w1t, const bf16* __restrict__ w2t,
        const bf16* __restrict__ w3t,
        const float* __restrict__ b1, const float* __restrict__ g1, const float* __restrict__ be1,
        const float* __restrict__ b2, const float* __restrict__ g2, const float* __restrict__ be2,
        const float* __restrict__ b3,
        const int* __restrict__ counts, const int* __restrict__ idxlist,
        const float* __restrict__ wlist,
        float* __restrict__ out) {
    __shared__ float HF[TM * D_HID];                        // 64 KB fp32 staging
    __shared__ __align__(32) unsigned short HB[TM * D_HID]; // 32 KB bf16 activations
    __shared__ float red1[TM * 8], red2[TM * 8], smu[TM], srs[TM];
    __shared__ int   s_tok[TM];
    __shared__ float s_w[TM];

    const int e    = blockIdx.y;
    const int tile = blockIdx.x;
    const int cnt  = counts[e];
    if (tile * TM >= cnt) return;      // block-uniform exit keeps EXEC full for WMMA

    const int tid  = threadIdx.x;
    const int wv   = tid >> 5;         // 0..7
    const int lane = tid & 31;
    const int half = lane >> 4;        // K-group select within fragments
    const int nl   = lane & 15;        // N column within 16-wide tile
    const int m    = lane & 15;        // A row (within 16-row sub-tile) this lane feeds

    if (tid < TM) {
        int gidx = tile * TM + tid;
        s_tok[tid] = (gidx < cnt) ? idxlist[(size_t)e * N_TOK + gidx] : -1;
        s_w[tid]   = (gidx < cnt) ? wlist[(size_t)e * N_TOK + gidx] : 0.0f;
    }
    __syncthreads();

    // ---------- stage 1: H1 = x_tile(32x256) @ W1(256x512) ----------
    {
        int tok0 = s_tok[m]      < 0 ? 0 : s_tok[m];
        int tok1 = s_tok[16 + m] < 0 ? 0 : s_tok[16 + m];
        const bf16* arow0 = xb + (size_t)tok0 * D_IN;
        const bf16* arow1 = xb + (size_t)tok1 * D_IN;
        v8f acc0[4] = {}, acc1[4] = {};
        for (int k = 0; k < D_IN / 32; ++k) {   // 8 K-steps
            union { v16bf v; v8bf h[2]; } a0, a1;
            a0.h[0] = *(const v8bf*)(arow0 + k * 32 +      8 * half);
            a0.h[1] = *(const v8bf*)(arow0 + k * 32 + 16 + 8 * half);
            a1.h[0] = *(const v8bf*)(arow1 + k * 32 +      8 * half);
            a1.h[1] = *(const v8bf*)(arow1 + k * 32 + 16 + 8 * half);
#pragma unroll
            for (int t = 0; t < 4; ++t) {
                int n = (wv * 4 + t) * 16 + nl;
                const bf16* bp = w1t + ((size_t)e * D_HID + n) * D_IN + k * 32;
                union { v16bf v; v8bf h[2]; } bu;
                bu.h[0] = *(const v8bf*)(bp +      8 * half);
                bu.h[1] = *(const v8bf*)(bp + 16 + 8 * half);
                v16bf b = bu.v;
                acc0[t] = __builtin_amdgcn_wmma_f32_16x16x32_bf16(
                              false, a0.v, false, b, (short)0, acc0[t], false, false);
                acc1[t] = __builtin_amdgcn_wmma_f32_16x16x32_bf16(
                              false, a1.v, false, b, (short)0, acc1[t], false, false);
            }
        }
#pragma unroll
        for (int t = 0; t < 4; ++t) {
            int n = (wv * 4 + t) * 16 + nl;
            float bias = b1[(size_t)e * D_HID + n];
#pragma unroll
            for (int j = 0; j < 8; ++j) {
                HF[(8 * half + j) * D_HID + n]        = acc0[t][j] + bias;
                HF[(16 + 8 * half + j) * D_HID + n]   = acc1[t][j] + bias;
            }
        }
    }
    ln_gelu_rows(HF, HB, g1 + (size_t)e * D_HID, be1 + (size_t)e * D_HID,
                 red1, red2, smu, srs, tid);

    // ---------- stage 2: H2 = H1b(32x512) @ W2(512x512) ----------
    {
        const bf16* hb = (const bf16*)HB;
        v8f acc0[4] = {}, acc1[4] = {};
        for (int k = 0; k < D_HID / 32; ++k) {  // 16 K-steps
            union { v16bf v; v8bf h[2]; } a0, a1;
            a0.h[0] = *(const v8bf*)(hb + m * D_HID + k * 32 +      8 * half);
            a0.h[1] = *(const v8bf*)(hb + m * D_HID + k * 32 + 16 + 8 * half);
            a1.h[0] = *(const v8bf*)(hb + (16 + m) * D_HID + k * 32 +      8 * half);
            a1.h[1] = *(const v8bf*)(hb + (16 + m) * D_HID + k * 32 + 16 + 8 * half);
#pragma unroll
            for (int t = 0; t < 4; ++t) {
                int n = (wv * 4 + t) * 16 + nl;
                const bf16* bp = w2t + ((size_t)e * D_HID + n) * D_HID + k * 32;
                union { v16bf v; v8bf h[2]; } bu;
                bu.h[0] = *(const v8bf*)(bp +      8 * half);
                bu.h[1] = *(const v8bf*)(bp + 16 + 8 * half);
                v16bf b = bu.v;
                acc0[t] = __builtin_amdgcn_wmma_f32_16x16x32_bf16(
                              false, a0.v, false, b, (short)0, acc0[t], false, false);
                acc1[t] = __builtin_amdgcn_wmma_f32_16x16x32_bf16(
                              false, a1.v, false, b, (short)0, acc1[t], false, false);
            }
        }
        __syncthreads();                        // all waves done reading HB
#pragma unroll
        for (int t = 0; t < 4; ++t) {
            int n = (wv * 4 + t) * 16 + nl;
            float bias = b2[(size_t)e * D_HID + n];
#pragma unroll
            for (int j = 0; j < 8; ++j) {
                HF[(8 * half + j) * D_HID + n]        = acc0[t][j] + bias;
                HF[(16 + 8 * half + j) * D_HID + n]   = acc1[t][j] + bias;
            }
        }
    }
    ln_gelu_rows(HF, HB, g2 + (size_t)e * D_HID, be2 + (size_t)e * D_HID,
                 red1, red2, smu, srs, tid);

    // ---------- stage 3: Y = H2b(32x512) @ W3(512x256), scatter-add ----------
    {
        const bf16* hb = (const bf16*)HB;
        v8f acc0[2] = {}, acc1[2] = {};
        for (int k = 0; k < D_HID / 32; ++k) {  // 16 K-steps
            union { v16bf v; v8bf h[2]; } a0, a1;
            a0.h[0] = *(const v8bf*)(hb + m * D_HID + k * 32 +      8 * half);
            a0.h[1] = *(const v8bf*)(hb + m * D_HID + k * 32 + 16 + 8 * half);
            a1.h[0] = *(const v8bf*)(hb + (16 + m) * D_HID + k * 32 +      8 * half);
            a1.h[1] = *(const v8bf*)(hb + (16 + m) * D_HID + k * 32 + 16 + 8 * half);
#pragma unroll
            for (int t = 0; t < 2; ++t) {
                int n = (wv * 2 + t) * 16 + nl;
                const bf16* bp = w3t + ((size_t)e * D_OUT + n) * D_HID + k * 32;
                union { v16bf v; v8bf h[2]; } bu;
                bu.h[0] = *(const v8bf*)(bp +      8 * half);
                bu.h[1] = *(const v8bf*)(bp + 16 + 8 * half);
                v16bf b = bu.v;
                acc0[t] = __builtin_amdgcn_wmma_f32_16x16x32_bf16(
                              false, a0.v, false, b, (short)0, acc0[t], false, false);
                acc1[t] = __builtin_amdgcn_wmma_f32_16x16x32_bf16(
                              false, a1.v, false, b, (short)0, acc1[t], false, false);
            }
        }
#pragma unroll
        for (int t = 0; t < 2; ++t) {
            int n = (wv * 2 + t) * 16 + nl;
            float bias = b3[(size_t)e * D_OUT + n];
#pragma unroll
            for (int j = 0; j < 8; ++j) {
                int mm0 = 8 * half + j;
                int mm1 = 16 + 8 * half + j;
                int tk0 = s_tok[mm0];
                int tk1 = s_tok[mm1];
                if (tk0 >= 0) {
                    float val = (acc0[t][j] + bias) * s_w[mm0];
                    __hip_atomic_fetch_add(out + (size_t)tk0 * D_OUT + n, val,
                                           __ATOMIC_RELAXED, __HIP_MEMORY_SCOPE_AGENT);
                }
                if (tk1 >= 0) {
                    float val = (acc1[t][j] + bias) * s_w[mm1];
                    __hip_atomic_fetch_add(out + (size_t)tk1 * D_OUT + n, val,
                                           __ATOMIC_RELAXED, __HIP_MEMORY_SCOPE_AGENT);
                }
            }
        }
    }
}

// ---------------- host-side launch ----------------
extern "C" void kernel_launch(void* const* d_in, const int* in_sizes, int n_in,
                              void* d_out, int out_size, void* d_ws, size_t ws_size,
                              hipStream_t stream) {
    const float* x   = (const float*)d_in[0];
    const float* gw1 = (const float*)d_in[1];
    const float* gw2 = (const float*)d_in[2];
    const float* W1  = (const float*)d_in[3];
    const float* b1  = (const float*)d_in[4];
    const float* g1  = (const float*)d_in[5];
    const float* be1 = (const float*)d_in[6];
    const float* W2  = (const float*)d_in[7];
    const float* b2  = (const float*)d_in[8];
    const float* g2  = (const float*)d_in[9];
    const float* be2 = (const float*)d_in[10];
    const float* W3  = (const float*)d_in[11];
    const float* b3  = (const float*)d_in[12];
    float* out = (float*)d_out;

    char* ws = (char*)d_ws;
    bf16* xb      = (bf16*)(ws + WS_XB);
    bf16* w1t     = (bf16*)(ws + WS_W1T);
    bf16* w2t     = (bf16*)(ws + WS_W2T);
    bf16* w3t     = (bf16*)(ws + WS_W3T);
    int*  counts  = (int*)(ws + WS_CNT);
    int*  idxlist = (int*)(ws + WS_IDX);
    float* wlist  = (float*)(ws + WS_WGT);

    // 1) bf16 transposed weights
    moe_prep_w<<<(R1 + R2 + R3) / 256, 256, 0, stream>>>(W1, W2, W3, w1t, w2t, w3t);
    // 2) zero output + counters
    moe_zero<<<(N_TOK * D_OUT) / 256, 256, 0, stream>>>(out, counts);
    // 3) gating / routing / x->bf16 (one wave32 per token, 8 waves per block)
    moe_gate<<<N_TOK / 8, 256, 0, stream>>>(x, gw1, gw2, xb, counts, idxlist, wlist);
    // 4) fused routed expert FFN (32-token tiles, B-fragment reuse x2)
    dim3 grid(N_TOK / TM, NEXP);
    moe_ffn<<<grid, 256, 0, stream>>>(xb, w1t, w2t, w3t,
                                      b1, g1, be1, b2, g2, be2, b3,
                                      counts, idxlist, wlist, out);
}
